// AsrModel_73684458930756
// MI455X (gfx1250) — compile-verified, matrix operand order checked by reading, not run on hardware
//
#include <hip/hip_runtime.h>
#include <math.h>

#define H    304
#define E    304
#define Vv   100
#define Bb   64
#define Ff   512
#define Pp   80
#define TENC (Bb * Ff)   // 32768
#define TDEC (Bb * Pp)   // 5120
#define G3   (3 * H)     // 912
#define KP   320         // K (=304) padded to a multiple of 32

typedef __attribute__((ext_vector_type(16))) _Float16 v16h;
typedef __attribute__((ext_vector_type(8)))  _Float16 v8h;   // 16 bytes
typedef __attribute__((ext_vector_type(8)))  float    v8f;

union V16H { v16h v; _Float16 h[16]; v8h q[2]; };

// ---------------------------------------------------------------------------
// Pack f32 [rows, cols] -> f16 [rowsP, KP] with zero padding. One-time
// conversion so the WMMA loop below is pure b128 loads + v_wmma.
// ---------------------------------------------------------------------------
__global__ void pack_f16(const float* __restrict__ src, int rows, int cols,
                         _Float16* __restrict__ dst, int rowsP)
{
    const int idx = blockIdx.x * blockDim.x + threadIdx.x;
    const int total = rowsP * KP;
    if (idx >= total) return;
    const int r = idx / KP, c = idx - r * KP;
    dst[idx] = (r < rows && c < cols) ? (_Float16)src[(size_t)r * cols + c]
                                      : (_Float16)0.0f;
}

// ---------------------------------------------------------------------------
// Out[m][n] = sum_k A[m][k] * W[n][k] + bias[n], A/W pre-packed f16 with
// row stride KP (rows padded to >= 16-tile). One wave32 per 16x16 tile:
// 4x global_load_b128 + 1x v_wmma_f32_16x16x32_f16 per 32-wide k-step,
// fully unrolled (KP/32 = 10 WMMAs per tile). No predication in the loop.
// ---------------------------------------------------------------------------
__global__ void wmma_gemm_f16_packed(const _Float16* __restrict__ A,  // [Mp,KP]
                                     const _Float16* __restrict__ W,  // [Np,KP]
                                     const float* __restrict__ bias,
                                     float* __restrict__ Out, int ldo,
                                     int M, int N)
{
    const int lane  = threadIdx.x & 31;
    const int gwave = (blockIdx.x * blockDim.x + threadIdx.x) >> 5;
    const int tilesN = (N + 15) >> 4;
    const int tilesM = (M + 15) >> 4;
    if (gwave >= tilesM * tilesN) return;     // wave-uniform exit
    const int tm = (gwave / tilesN) << 4;
    const int tn = (gwave % tilesN) << 4;

    const int half = lane >> 4;               // 0: K 0-7,16-23; 1: K 8-15,24-31
    const int l16  = lane & 15;
    // A-frag: lane holds row tm+l16, halfs[0..7]=K kA.., halfs[8..15]=K kA+16..
    const _Float16* ap = A + (size_t)(tm + l16) * KP + half * 8;
    const _Float16* wp = W + (size_t)(tn + l16) * KP + half * 8;

    v8f acc = {};
#pragma unroll
    for (int k0 = 0; k0 < KP; k0 += 32) {
        V16H a, b;
        a.q[0] = *(const v8h*)(ap + k0);
        a.q[1] = *(const v8h*)(ap + k0 + 16);
        b.q[0] = *(const v8h*)(wp + k0);
        b.q[1] = *(const v8h*)(wp + k0 + 16);
        acc = __builtin_amdgcn_wmma_f32_16x16x32_f16(
                  false, a.v, false, b.v, (short)0, acc, false, false);
    }

    // C/D layout: VGPR v holds M = tm + v + 8*half, N = tn + l16
    const int n = tn + l16;
    if (n < N) {
        const float bv = bias ? bias[n] : 0.0f;
        const int mbase = tm + (half << 3);
#pragma unroll
        for (int v = 0; v < 8; ++v) {
            const int m = mbase + v;
            if (m < M) Out[(size_t)m * ldo + n] = acc[v] + bv;
        }
    }
}

// ---------------------------------------------------------------------------
// Serial GRU scan: one persistent 1024-thread workgroup (32 wave32 waves on
// one WGP). h in LDS (broadcast reads); gate vector in LDS; Whh (1.1 MB f32)
// streamed from global and served from the 192 MB L2 every step. If ysf16 is
// given, hidden states are emitted as packed f16 rows [T, KP] with the pad
// columns zeroed (ready to be the A-matrix of the logits WMMA GEMM).
// ---------------------------------------------------------------------------
__global__ __launch_bounds__(1024) void gru_scan_kernel(
    const float* __restrict__ xp,    // [T, 912]  x @ Wih^T + bih
    const float* __restrict__ Whh,   // [912, 304]
    const float* __restrict__ bhh,   // [912]
    const float* __restrict__ h0,    // [304] or nullptr -> zeros
    float* __restrict__ hT,          // [304]
    _Float16* __restrict__ ysf16,    // [T, KP] or nullptr
    int T)
{
    __shared__ float h[H];
    __shared__ float gg[G3];
    const int tid = threadIdx.x;
    if (tid < H) h[tid] = h0 ? h0[tid] : 0.0f;
    __syncthreads();

    const float* wrow = Whh + (size_t)tid * H;
    const float bj = (tid < G3) ? bhh[tid] : 0.0f;

    for (int t = 0; t < T; ++t) {
        if (tid < G3) {                       // hg = h @ Whh^T + bhh
            float g = bj;
#pragma unroll 8
            for (int k = 0; k < H; ++k) g += wrow[k] * h[k];
            gg[tid] = g;
        }
        __syncthreads();
        if (tid < H) {                        // gates + state update
            const float* xg = xp + (size_t)t * G3;
            const float xr = xg[tid], xz = xg[H + tid], xn = xg[2 * H + tid];
            const float hr = gg[tid], hz = gg[H + tid], hn = gg[2 * H + tid];
            const float r = 1.0f / (1.0f + __expf(-(xr + hr)));
            const float z = 1.0f / (1.0f + __expf(-(xz + hz)));
            const float n = tanhf(xn + r * hn);
            const float hnew = (1.0f - z) * n + z * h[tid];
            h[tid] = hnew;                    // only thread tid touches h[tid]
            if (ysf16) ysf16[(size_t)t * KP + tid] = (_Float16)hnew;
        } else if (tid < KP) {                // zero the f16 pad columns
            if (ysf16) ysf16[(size_t)t * KP + tid] = (_Float16)0.0f;
        }
        __syncthreads();
    }
    if (tid < H) hT[tid] = h[tid];
}

// ---------------------------------------------------------------------------
// Decoder token roll (teacher forcing) + embedding gather, written directly
// as the packed f16 A-matrix [TDEC, KP] (pad columns zeroed); also emits
// target_cal as floats.
// ---------------------------------------------------------------------------
__global__ void build_dec_inputs_f16(const int* __restrict__ target,  // [5120]
                                     const float* __restrict__ emb,   // [100,304]
                                     _Float16* __restrict__ pA,       // [5120,KP]
                                     float* __restrict__ target_cal)  // [5120]
{
    const int idx = blockIdx.x * blockDim.x + threadIdx.x;
    const int total = TDEC * KP;
    if (idx >= total) return;
    const int i = idx / KP, c = idx - i * KP;
    const int ft = target[i];
    int tk;
    if ((i % Pp) == 0) tk = (i == 0) ? 0 : target[i - 1];
    else               tk = ft;
    pA[idx] = (c < E) ? (_Float16)emb[(size_t)tk * E + c] : (_Float16)0.0f;
    if (c == 0) target_cal[i] = (float)ft;
}

// ---------------------------------------------------------------------------
// Per-row softmax over V=100 logits + first-index argmax. One block per row,
// wave32-safe LDS tree reductions.
// ---------------------------------------------------------------------------
__global__ __launch_bounds__(128) void softmax_argmax_kernel(
    const float* __restrict__ logits,  // [TDEC, Vv]
    float* __restrict__ sm_out,        // [TDEC, Vv]
    float* __restrict__ arg_out)       // [TDEC]
{
    __shared__ float red[128];
    __shared__ int   redi[128];
    const int row = blockIdx.x;
    const int tid = threadIdx.x;
    const float v = (tid < Vv) ? logits[(size_t)row * Vv + tid] : -INFINITY;

    red[tid] = v; __syncthreads();
    for (int s = 64; s > 0; s >>= 1) {
        if (tid < s) red[tid] = fmaxf(red[tid], red[tid + s]);
        __syncthreads();
    }
    const float mx = red[0]; __syncthreads();

    const float e = (tid < Vv) ? __expf(v - mx) : 0.0f;
    red[tid] = e; __syncthreads();
    for (int s = 64; s > 0; s >>= 1) {
        if (tid < s) red[tid] += red[tid + s];
        __syncthreads();
    }
    const float inv = 1.0f / red[0]; __syncthreads();
    if (tid < Vv) sm_out[(size_t)row * Vv + tid] = e * inv;

    red[tid] = v; redi[tid] = tid; __syncthreads();
    for (int s = 64; s > 0; s >>= 1) {
        if (tid < s) {
            const float a = red[tid], b = red[tid + s];
            const int   ia = redi[tid], ib = redi[tid + s];
            if (b > a || (b == a && ib < ia)) { red[tid] = b; redi[tid] = ib; }
        }
        __syncthreads();
    }
    if (tid == 0) arg_out[row] = (float)redi[0];
}

// ---------------------------------------------------------------------------
extern "C" void kernel_launch(void* const* d_in, const int* in_sizes, int n_in,
                              void* d_out, int out_size, void* d_ws, size_t ws_size,
                              hipStream_t stream)
{
    const float* input   = (const float*)d_in[0];
    const int*   target  = (const int*)  d_in[1];
    const float* emb     = (const float*)d_in[2];
    const float* enc_Wih = (const float*)d_in[3];
    const float* enc_Whh = (const float*)d_in[4];
    const float* enc_bih = (const float*)d_in[5];
    const float* enc_bhh = (const float*)d_in[6];
    const float* dec_Wih = (const float*)d_in[7];
    const float* dec_Whh = (const float*)d_in[8];
    const float* dec_bih = (const float*)d_in[9];
    const float* dec_bhh = (const float*)d_in[10];
    const float* lin_W   = (const float*)d_in[11];
    const float* lin_b   = (const float*)d_in[12];

    float* out     = (float*)d_out;
    float* sm_out  = out;                       // [5120*100] softmax_cal
    float* tc_out  = out + (size_t)TDEC * Vv;   // [5120]     target_cal
    float* arg_out = tc_out + TDEC;             // [5120]     asr_outputs

    // ---- workspace layout -------------------------------------------------
    float* ws     = (float*)d_ws;
    float* xp_enc = ws;                                  // 32768*912 f32
    float* xp_dec = xp_enc + (size_t)TENC * G3;          // 5120*912  f32
    float* logits = xp_dec + (size_t)TDEC * G3;          // 5120*100  f32
    float* enc_hT = logits + (size_t)TDEC * Vv;          // 304       f32
    _Float16* f16ws   = (_Float16*)(enc_hT + H);
    _Float16* pA_enc  = f16ws;                           // 32768*KP
    _Float16* pW_eih  = pA_enc + (size_t)TENC * KP;      // 912*KP
    _Float16* pW_dih  = pW_eih + (size_t)G3 * KP;        // 912*KP
    _Float16* pW_lin  = pW_dih + (size_t)G3 * KP;        // 112*KP (100 padded)
    _Float16* pA_dec  = pW_lin + (size_t)112 * KP;       // 5120*KP
    _Float16* pHs     = pA_dec + (size_t)TDEC * KP;      // 5120*KP

    auto pack = [&](const float* src, int rows, int cols, _Float16* dst, int rowsP) {
        const int total = rowsP * KP;
        pack_f16<<<(total + 255) / 256, 256, 0, stream>>>(src, rows, cols, dst, rowsP);
    };
    auto gemm = [&](const _Float16* A, const _Float16* W, const float* bias,
                    float* O, int ldo, int M, int N) {
        const int tiles  = ((M + 15) / 16) * ((N + 15) / 16);
        const int blocks = (tiles + 7) / 8;              // 8 wave32 per block
        wmma_gemm_f16_packed<<<blocks, 256, 0, stream>>>(A, W, bias, O, ldo, M, N);
    };

    // 0) one-time f32 -> packed-f16 conversions
    pack(input,   TENC, E, pA_enc, TENC);
    pack(enc_Wih, G3,   E, pW_eih, G3);
    pack(dec_Wih, G3,   E, pW_dih, G3);
    pack(lin_W,   Vv,   H, pW_lin, 112);

    // 1) encoder input projection: xp_enc = input @ enc_Wih^T + enc_bih
    gemm(pA_enc, pW_eih, enc_bih, xp_enc, G3, TENC, G3);

    // 2) encoder serial scan (T = 32768), h0 = 0
    gru_scan_kernel<<<1, 1024, 0, stream>>>(xp_enc, enc_Whh, enc_bhh,
                                            (const float*)nullptr, enc_hT,
                                            (_Float16*)nullptr, TENC);

    // 3) decoder teacher-forced inputs (packed f16) + target_cal output
    {
        const int total = TDEC * KP;
        build_dec_inputs_f16<<<(total + 255) / 256, 256, 0, stream>>>(
            target, emb, pA_dec, tc_out);
    }

    // 4) decoder input projection: xp_dec = dec_xs @ dec_Wih^T + dec_bih
    gemm(pA_dec, pW_dih, dec_bih, xp_dec, G3, TDEC, G3);

    // 5) decoder serial scan (T = 5120), h0 = enc_hT; emit packed-f16 hs
    gru_scan_kernel<<<1, 1024, 0, stream>>>(xp_dec, dec_Whh, dec_bhh,
                                            enc_hT, enc_hT, pHs, TDEC);

    // 6) logits = hs @ lin_W^T + lin_b
    gemm(pHs, pW_lin, lin_b, logits, Vv, TDEC, Vv);

    // 7) softmax + argmax -> d_out
    softmax_argmax_kernel<<<TDEC, 128, 0, stream>>>(logits, sm_out, arg_out);
}
